// GAT_58514634441267
// MI455X (gfx1250) — compile-verified, matrix operand order checked by reading, not run on hardware
//
#include <hip/hip_runtime.h>
#include <hip/hip_bf16.h>
#include <stdint.h>

// ---------------------------------------------------------------------------
// GAT net on MI455X (gfx1250). Dense layers -> v_wmma_f32_16x16x32_bf16 with
// 4x M register blocking: wave = 64x16 output strip, B fragment loaded once
// per K-step and reused by 4 WMMAs (1.25 KB moved per 16 KFLOP WMMA).
// All operands pre-packed bf16 so fragments are contiguous global_load_b128
// (ISA 7.12.2: a lane's 16 elements = two contiguous 8-elem K-runs).
// Segment softmax/aggregation via L2 float atomics. Workspace ~260 MB.
// ---------------------------------------------------------------------------

#define NN_  20000
#define NE_  64000
#define B_   2
#define L_   128

typedef __attribute__((ext_vector_type(16))) __bf16    v16bf;
typedef __attribute__((ext_vector_type(8)))  float     v8f;
typedef __attribute__((ext_vector_type(4)))  unsigned  u32x4;

union frag16 { v16bf v; u32x4 u[2]; };

__device__ inline __bf16 f2bf(float f) {
  unsigned u = __float_as_uint(f);
  unsigned r = u + 0x7fffu + ((u >> 16) & 1u);
  unsigned short s = (unsigned short)(r >> 16);
  __bf16 b; __builtin_memcpy(&b, &s, 2); return b;
}
__device__ inline float bf2f(__bf16 b) {
  unsigned short s; __builtin_memcpy(&s, &b, 2);
  return __uint_as_float((unsigned)s << 16);
}

// ---- A-side gather: resolve(row) hoists indirections out of the K loop ----
struct SegPlain {                       // bf16 row-major [M, ld]
  const __bf16* __restrict__ p; int ld;
  struct Row { const __bf16* b; };
  __device__ Row resolve(int row) const { return { p + (long)row * ld }; }
  __device__ const __bf16* at(const Row& r, int kr) const { return r.b + kr; }
};
struct SegEdgeCat {                     // [nl[snd] | nl[rcv] | el], bf16 [*,128]
  const __bf16* __restrict__ nl; const __bf16* __restrict__ el;
  const int* __restrict__ snd; const int* __restrict__ rcv;
  struct Row { const __bf16* b0; const __bf16* b1; const __bf16* b2; };
  __device__ Row resolve(int row) const {
    int b = row / NE_; int e = row - b * NE_;
    return { nl + ((long)b * NN_ + snd[e]) * L_,
             nl + ((long)b * NN_ + rcv[e]) * L_,
             el + (long)row * L_ };
  }
  __device__ const __bf16* at(const Row& r, int kr) const {
    return (kr < L_) ? r.b0 + kr
         : (kr < 2 * L_) ? r.b1 + (kr - L_) : r.b2 + (kr - 2 * L_);
  }
};
struct SegNodeCat {                     // [nl | agg], bf16 [*,128]
  const __bf16* __restrict__ nl; const __bf16* __restrict__ agg;
  struct Row { const __bf16* b0; const __bf16* b1; };
  __device__ Row resolve(int row) const {
    return { nl + (long)row * L_, agg + (long)row * L_ };
  }
  __device__ const __bf16* at(const Row& r, int kr) const {
    return (kr < L_) ? r.b0 + kr : r.b1 + (kr - L_);
  }
};

// ---- WMMA GEMM: wave = 64x16 strip (4 M-subtiles share the B fragment) ----
template <typename S>
__global__ __launch_bounds__(256) void k_gemm(S seg, const __bf16* __restrict__ Wt,
                                              int KpW, const float* __restrict__ bias,
                                              void* __restrict__ outp,
                                              int M, int N, int Kp, int relu,
                                              int out_is_f32) {
  int wave = threadIdx.x >> 5, lane = threadIdx.x & 31;
  int m0 = blockIdx.x * 64;
  int n0 = (blockIdx.y * 8 + wave) * 16;
  if (n0 >= N) return;                       // wave-uniform; EXEC all-ones for WMMA
  int lh8 = (lane >> 4) << 3;
  typename S::Row rows[4];
#pragma unroll
  for (int t = 0; t < 4; ++t) {
    int r = m0 + t * 16 + (lane & 15);
    if (r >= M) r = M - 1;                   // never taken (M % 64 == 0), safety
    rows[t] = seg.resolve(r);
  }
  const __bf16* wrow = Wt + (long)(n0 + (lane & 15)) * KpW;  // Npad=ceil16(N) rows
  v8f acc[4] = {{}, {}, {}, {}};
  for (int k0 = 0; k0 < Kp; k0 += 32) {
    frag16 b;
    b.u[0] = *(const u32x4*)(wrow + k0 + lh8);
    b.u[1] = *(const u32x4*)(wrow + k0 + 16 + lh8);
#pragma unroll
    for (int t = 0; t < 4; ++t) {
      frag16 a;
      a.u[0] = *(const u32x4*)seg.at(rows[t], k0 + lh8);
      a.u[1] = *(const u32x4*)seg.at(rows[t], k0 + 16 + lh8);
      acc[t] = __builtin_amdgcn_wmma_f32_16x16x32_bf16(false, a.v, false, b.v,
                                                       (short)0, acc[t], false, false);
    }
  }
  int nc = n0 + (lane & 15);
  float bv = (nc < N) ? bias[nc] : 0.f;
#pragma unroll
  for (int t = 0; t < 4; ++t) {
    int mbase = m0 + t * 16 + ((lane >> 4) << 3);
#pragma unroll
    for (int r = 0; r < 8; ++r) {
      int m = mbase + r;
      if (m < M && nc < N) {
        float v = acc[t][r] + bv;
        if (relu) v = fmaxf(v, 0.f);
        long idx = (long)m * N + nc;
        if (out_is_f32) ((float*)outp)[idx] = v;
        else            ((__bf16*)outp)[idx] = f2bf(v);
      }
    }
  }
}

// ---------------- packing kernels (once per launch, tiny) -------------------
__global__ __launch_bounds__(256) void k_pack_w(const float* __restrict__ W,
                                                __bf16* __restrict__ Wt,
                                                int K, int N, int Kp, long total) {
  long i = (long)blockIdx.x * 256 + threadIdx.x;     // over Npad*Kp
  if (i >= total) return;
  int n = (int)(i / Kp), kp = (int)(i - (long)n * Kp);
  Wt[i] = f2bf((n < N && kp < K) ? W[(long)kp * N + n] : 0.f);
}
__global__ __launch_bounds__(256) void k_pack_nodeenc(const float* __restrict__ nodef,
                                                      const float* __restrict__ imgf,
                                                      __bf16* __restrict__ nf) {
  long i = (long)blockIdx.x * 256 + threadIdx.x;     // over B*NN*544
  if (i >= (long)B_ * NN_ * 544) return;
  int row = (int)(i / 544), k = (int)(i - (long)row * 544);
  int b = row / NN_, n = row - b * NN_;
  float v = (k < 3) ? nodef[n * 3 + k] : ((k < 515) ? imgf[b * 512 + (k - 3)] : 0.f);
  nf[i] = f2bf(v);
}
__global__ __launch_bounds__(256) void k_pack_edgeenc(const float* __restrict__ edgef,
                                                      __bf16* __restrict__ ef) {
  long i = (long)blockIdx.x * 256 + threadIdx.x;     // over B*NE*32
  if (i >= (long)B_ * NE_ * 32) return;
  int row = (int)(i / 32), k = (int)(i - (long)row * 32);
  int e = row % NE_;
  ef[i] = f2bf((k < 4) ? edgef[e * 4 + k] : 0.f);
}
__global__ __launch_bounds__(256) void k_f32_to_bf(const float* __restrict__ src,
                                                   __bf16* __restrict__ dst, long n) {
  long i = (long)blockIdx.x * 256 + threadIdx.x;
  if (i < n) dst[i] = f2bf(src[i]);
}

// ---------------- layernorm (wave per 128-feature row), bf16 in/out ---------
__global__ __launch_bounds__(256) void k_layernorm(const __bf16* __restrict__ x,
                                                   const float* __restrict__ g,
                                                   const float* __restrict__ bb,
                                                   __bf16* __restrict__ out,
                                                   int M, int accumulate) {
  int row = blockIdx.x * 8 + (threadIdx.x >> 5);
  if (row >= M) return;
  int lane = threadIdx.x & 31;
  float v[4]; float s = 0.f;
#pragma unroll
  for (int i = 0; i < 4; ++i) { v[i] = bf2f(x[(long)row * L_ + lane + 32 * i]); s += v[i]; }
#pragma unroll
  for (int o = 16; o > 0; o >>= 1) s += __shfl_xor(s, o, 32);
  float mean = s * (1.f / L_);
  float vs = 0.f;
#pragma unroll
  for (int i = 0; i < 4; ++i) { float d = v[i] - mean; vs += d * d; }
#pragma unroll
  for (int o = 16; o > 0; o >>= 1) vs += __shfl_xor(vs, o, 32);
  float rinv = rsqrtf(vs * (1.f / L_) + 1e-5f);
#pragma unroll
  for (int i = 0; i < 4; ++i) {
    int c = lane + 32 * i;
    float o = (v[i] - mean) * rinv * g[c] + bb[c];
    long idx = (long)row * L_ + c;
    out[idx] = f2bf(accumulate ? (bf2f(out[idx]) + o) : o);
  }
}

// ---------------- attention ---------------------------------------------------
__device__ inline unsigned fenc(float f) {
  unsigned u = __float_as_uint(f);
  return (u & 0x80000000u) ? ~u : (u | 0x80000000u);
}
__device__ inline float fdec(unsigned u) {
  return (u & 0x80000000u) ? __uint_as_float(u & 0x7fffffffu) : __uint_as_float(~u);
}

__global__ __launch_bounds__(256) void k_att_logits(const __bf16* __restrict__ e_new,
                                                    const float* __restrict__ Wa,
                                                    const float* __restrict__ ba,
                                                    const int* __restrict__ rcv,
                                                    float* __restrict__ logits,
                                                    unsigned* __restrict__ maxb, int M) {
  int row = blockIdx.x * 8 + (threadIdx.x >> 5);
  if (row >= M) return;
  int lane = threadIdx.x & 31;
  float s = 0.f;
#pragma unroll
  for (int i = 0; i < 4; ++i)
    s += bf2f(e_new[(long)row * L_ + lane + 32 * i]) * Wa[lane + 32 * i];
#pragma unroll
  for (int o = 16; o > 0; o >>= 1) s += __shfl_xor(s, o, 32);
  if (lane == 0) {
    float l = s + ba[0];
    l = (l > 0.f) ? l : 0.2f * l;
    logits[row] = l;
    int b = row / NE_; int e = row - b * NE_;
    atomicMax(&maxb[b * NN_ + rcv[e]], fenc(l));
  }
}
__global__ __launch_bounds__(256) void k_att_exp(float* __restrict__ exb,
                                                 const unsigned* __restrict__ maxb,
                                                 float* __restrict__ sumb,
                                                 const int* __restrict__ rcv, int M) {
  int row = blockIdx.x * blockDim.x + threadIdx.x;
  if (row >= M) return;
  int b = row / NE_; int e = row - b * NE_;
  int seg = b * NN_ + rcv[e];
  float ex = __expf(exb[row] - fdec(maxb[seg]));
  exb[row] = ex;
  atomicAdd(&sumb[seg], ex);
}
__global__ __launch_bounds__(128) void k_att_agg(const __bf16* __restrict__ e_new,
                                                 const float* __restrict__ exb,
                                                 const float* __restrict__ sumb,
                                                 const int* __restrict__ rcv,
                                                 float* __restrict__ agg) {
  int row = blockIdx.x;
  int b = row / NE_; int e = row - b * NE_;
  int seg = b * NN_ + rcv[e];
  float att = exb[row] / sumb[seg];
  atomicAdd(&agg[(long)seg * L_ + threadIdx.x],
            bf2f(e_new[(long)row * L_ + threadIdx.x]) * att);
}

// ---------------- utilities ---------------------------------------------------
__global__ __launch_bounds__(256) void k_fill_u32(unsigned* __restrict__ p, unsigned v, long n) {
  long i = (long)blockIdx.x * blockDim.x + threadIdx.x;
  if (i < n) p[i] = v;
}
__global__ __launch_bounds__(256) void k_add_bf(__bf16* __restrict__ dst,
                                                const __bf16* __restrict__ src, long n) {
  long i = (long)blockIdx.x * blockDim.x + threadIdx.x;
  if (i < n) dst[i] = f2bf(bf2f(dst[i]) + bf2f(src[i]));
}

// ---------------------------------------------------------------------------
extern "C" void kernel_launch(void* const* d_in, const int* in_sizes, int n_in,
                              void* d_out, int out_size, void* d_ws, size_t ws_size,
                              hipStream_t stream) {
  const float* nodef = (const float*)d_in[0];
  const float* edgef = (const float*)d_in[1];
  const float* imgf  = (const float*)d_in[2];
  const int*   snd   = (const int*)d_in[3];
  const int*   rcv   = (const int*)d_in[4];
  const float* const* P = (const float* const*)(d_in + 5);   // pytree order
  const int DEC = 54, EE = 60, EN = 66;

  const long MN = (long)B_ * NN_;       // 40000
  const long ME = (long)B_ * NE_;       // 128000

  // ---- workspace carve (all 16B aligned) ----
  char* w = (char*)d_ws;
  auto take = [&](long bytes) { char* p = w; w += (bytes + 15) & ~15L; return p; };
  __bf16* nf_bf = (__bf16*)take(MN * 544 * 2);
  __bf16* ef_bf = (__bf16*)take(ME * 32 * 2);
  __bf16* bufA  = (__bf16*)take(ME * 256 * 2);
  __bf16* bufB  = (__bf16*)take(ME * L_ * 2);
  __bf16* e_new = (__bf16*)take(ME * L_ * 2);
  __bf16* el    = (__bf16*)take(ME * L_ * 2);
  __bf16* nl    = (__bf16*)take(MN * L_ * 2);
  __bf16* aggbf = (__bf16*)take(MN * L_ * 2);
  float*  aggf  = (float*)take(MN * L_ * 4);
  float*  exb   = (float*)take(ME * 4);
  unsigned* maxb = (unsigned*)take((long)B_ * NN_ * 4);
  float*  sumb  = (float*)take((long)B_ * NN_ * 4);
  __bf16* wpool = (__bf16*)take(700000L * 2);

  dim3 blk(256);
  auto cdiv = [](long a, long b) { return (unsigned)((a + b - 1) / b); };
  auto ggrid = [&](long M, int N) { return dim3(cdiv(M, 64), cdiv(N, 128)); };

  // ---- pack weights: Wt bf16 [ceil16(N), Kp], transposed ----
  long woff = 0;
  auto packw = [&](const float* W, int K, int N, int Kp) {
    int Np = (N + 15) & ~15;
    __bf16* dst = wpool + woff;
    long total = (long)Np * Kp; woff += total;
    k_pack_w<<<dim3(cdiv(total, 256)), blk, 0, stream>>>(W, dst, K, N, Kp, total);
    return dst;
  };
  __bf16* Wen0 = packw(P[EN + 0], 515, 256, 544);
  __bf16* Wen1 = packw(P[EN + 2], 256, 128, 256);
  __bf16* Wee0 = packw(P[EE + 0], 4, 256, 32);
  __bf16* Wee1 = packw(P[EE + 2], 256, 128, 256);
  __bf16 *We[3][3], *Wn[3][3];
  for (int i = 0; i < 3; ++i) {
    const int PB = i * 18;
    We[i][0] = packw(P[PB + 2], 384, 128, 384);
    We[i][1] = packw(P[PB + 4], 128, 128, 128);
    We[i][2] = packw(P[PB + 6], 128, 128, 128);
    Wn[i][0] = packw(P[PB + 10], 256, 128, 256);
    Wn[i][1] = packw(P[PB + 12], 128, 128, 128);
    Wn[i][2] = packw(P[PB + 14], 128, 128, 128);
  }
  __bf16* Wd0 = packw(P[DEC + 0], 128, 128, 128);
  __bf16* Wd1 = packw(P[DEC + 2], 128, 128, 128);
  __bf16* Wd2 = packw(P[DEC + 4], 128, 3, 128);

  // ---- pack encoder inputs ----
  k_pack_nodeenc<<<dim3(cdiv(MN * 544, 256)), blk, 0, stream>>>(nodef, imgf, nf_bf);
  k_pack_edgeenc<<<dim3(cdiv(ME * 32, 256)), blk, 0, stream>>>(edgef, ef_bf);

  // ---- encoders ----
  k_gemm<<<ggrid(MN, 256), blk, 0, stream>>>(SegPlain{nf_bf, 544}, Wen0, 544,
      P[EN + 1], bufA, (int)MN, 256, 544, 1, 0);
  k_gemm<<<ggrid(MN, 128), blk, 0, stream>>>(SegPlain{bufA, 256}, Wen1, 256,
      P[EN + 3], bufB, (int)MN, 128, 256, 0, 0);
  k_layernorm<<<dim3(cdiv(MN, 8)), blk, 0, stream>>>(bufB, P[EN + 4], P[EN + 5], nl, (int)MN, 0);
  k_gemm<<<ggrid(ME, 256), blk, 0, stream>>>(SegPlain{ef_bf, 32}, Wee0, 32,
      P[EE + 1], bufA, (int)ME, 256, 32, 1, 0);
  k_gemm<<<ggrid(ME, 128), blk, 0, stream>>>(SegPlain{bufA, 256}, Wee1, 256,
      P[EE + 3], bufB, (int)ME, 128, 256, 0, 0);
  k_layernorm<<<dim3(cdiv(ME, 8)), blk, 0, stream>>>(bufB, P[EE + 4], P[EE + 5], el, (int)ME, 0);

  // ---- message-passing blocks ----
  for (int i = 0; i < 3; ++i) {
    const int PB = i * 18;
    k_fill_u32<<<dim3(cdiv(2L * B_ * NN_, 256)), blk, 0, stream>>>(maxb, 0u, 2L * B_ * NN_);
    k_fill_u32<<<dim3(cdiv(MN * L_, 256)), blk, 0, stream>>>((unsigned*)aggf, 0u, MN * L_);
    // edge MLP -> LN -> e_new
    k_gemm<<<ggrid(ME, 128), blk, 0, stream>>>(SegEdgeCat{nl, el, snd, rcv}, We[i][0], 384,
        P[PB + 3], bufA, (int)ME, 128, 384, 1, 0);
    k_gemm<<<ggrid(ME, 128), blk, 0, stream>>>(SegPlain{bufA, 128}, We[i][1], 128,
        P[PB + 5], bufB, (int)ME, 128, 128, 1, 0);
    k_gemm<<<ggrid(ME, 128), blk, 0, stream>>>(SegPlain{bufB, 128}, We[i][2], 128,
        P[PB + 7], bufA, (int)ME, 128, 128, 0, 0);
    k_layernorm<<<dim3(cdiv(ME, 8)), blk, 0, stream>>>(bufA, P[PB + 8], P[PB + 9],
        e_new, (int)ME, 0);
    // attention softmax over receivers + weighted scatter-add
    k_att_logits<<<dim3(cdiv(ME, 8)), blk, 0, stream>>>(e_new, P[PB + 0], P[PB + 1],
        rcv, exb, maxb, (int)ME);
    k_att_exp<<<dim3(cdiv(ME, 256)), blk, 0, stream>>>(exb, maxb, sumb, rcv, (int)ME);
    k_att_agg<<<dim3((unsigned)ME), dim3(128), 0, stream>>>(e_new, exb, sumb, rcv, aggf);
    k_f32_to_bf<<<dim3(cdiv(MN * L_, 256)), blk, 0, stream>>>(aggf, aggbf, MN * L_);
    // node MLP -> nl += LN(out)
    k_gemm<<<ggrid(MN, 128), blk, 0, stream>>>(SegNodeCat{nl, aggbf}, Wn[i][0], 256,
        P[PB + 11], bufA, (int)MN, 128, 256, 1, 0);
    k_gemm<<<ggrid(MN, 128), blk, 0, stream>>>(SegPlain{bufA, 128}, Wn[i][1], 128,
        P[PB + 13], bufB, (int)MN, 128, 128, 1, 0);
    k_gemm<<<ggrid(MN, 128), blk, 0, stream>>>(SegPlain{bufB, 128}, Wn[i][2], 128,
        P[PB + 15], bufA, (int)MN, 128, 128, 0, 0);
    k_layernorm<<<dim3(cdiv(MN, 8)), blk, 0, stream>>>(bufA, P[PB + 16], P[PB + 17],
        nl, (int)MN, 1);
    // el += e_new
    k_add_bf<<<dim3(cdiv(ME * L_, 256)), blk, 0, stream>>>(el, e_new, ME * L_);
  }

  // ---- decoder ----
  k_gemm<<<ggrid(MN, 128), blk, 0, stream>>>(SegPlain{nl, 128}, Wd0, 128,
      P[DEC + 1], bufA, (int)MN, 128, 128, 1, 0);
  k_gemm<<<ggrid(MN, 128), blk, 0, stream>>>(SegPlain{bufA, 128}, Wd1, 128,
      P[DEC + 3], bufB, (int)MN, 128, 128, 1, 0);
  k_gemm<<<ggrid(MN, 16), blk, 0, stream>>>(SegPlain{bufB, 128}, Wd2, 128,
      P[DEC + 5], d_out, (int)MN, 3, 128, 0, 1);
}
